// DGCRN_51900384805082
// MI455X (gfx1250) — matrix operand order repaired and checked by reading, not compiled, not run on hardware
//
#include <hip/hip_runtime.h>
#include <math.h>

// ---- problem constants (from reference) ----
#define BB 32
#define LL 12
#define NN 325
#define CIN 2
#define HID 64
#define NODE_DIM 40
#define HYP_DIM 16
#define MID_DIM 2
#define DIN 198          // (GDEP+1)*(HID+CIN)
#define AWc 0.05f
#define BWc 0.95f
#define GWc 0.95f
#define TAc 3.0f
#define XDIM 66          // CIN + HID

typedef __attribute__((ext_vector_type(16))) _Float16 v16h;
typedef __attribute__((ext_vector_type(8)))  float    v8f;

// =====================================================================
// Generic batched GEMM on WMMA f16->f32, templated on transpose/scale so
// the staging loops are branch-free straight-line code.
// Block = 128 threads (4 waves) -> 32x32 block tile; each wave owns a
// 16x16 quadrant (A rows shared by 2 waves, B cols shared by 2 waves).
//   C[m,n] = epi( alpha * sum_k Aop[m,k]*Bop[k,n] )
//   Aop[m,k] = (TA ? A[k*lda+m] : A[m*lda+k]) * (SC ? scaleK[k] : 1)
//   Bop[k,n] = (TB ? Bm[n*ldb+k] : Bm[k*ldb+n])
//   epi: + addScale*Cadd[m,n] + bias[n] + (accumulate ? C_old : 0), then act
//   act: 0 none, 1 sigmoid, 2 tanh.  blockIdx.z = batch.
// =====================================================================
template<int TA, int TB, int SC>
__global__ __launch_bounds__(128)
void k_gemm(const float* __restrict__ A, long sA, int lda,
            const float* __restrict__ scaleK, long sScale,
            const float* __restrict__ Bm, long sB, int ldb,
            float* __restrict__ C, long sC, int ldc,
            const float* __restrict__ Cadd, long sCadd, int ldcadd, float addScale,
            const float* __restrict__ bias,
            int M, int Nn, int K, float alpha, int act, int accumulate)
{
    __shared__ _Float16 lA[32 * 32];   // [m within tile][k]
    __shared__ _Float16 lB[32 * 32];   // [n within tile][k]  (K-contiguous)
    const int tid  = threadIdx.x;
    const int lane = tid & 31;
    const int wv   = tid >> 5;               // wave 0..3
    const int bm0  = blockIdx.x * 32;
    const int bn0  = blockIdx.y * 32;
    const int m0   = bm0 + (wv >> 1) * 16;   // this wave's 16x16 quadrant
    const int n0   = bn0 + (wv & 1) * 16;
    const int bz   = blockIdx.z;
    A  += (long)bz * sA;
    Bm += (long)bz * sB;
    C  += (long)bz * sC;
    if (Cadd) Cadd += (long)bz * sCadd;
    if (SC)   scaleK += (long)bz * sScale;

    const bool fullM = (bm0 + 32 <= M);
    const bool fullN = (bn0 + 32 <= Nn);

    v8f acc = {0.f,0.f,0.f,0.f,0.f,0.f,0.f,0.f};
    const int row = lane & 15;
    const int hh  = lane >> 4;
    const int aRow = ((wv >> 1) * 16 + row) << 5;   // lane's A row base in lA
    const int bRow = ((wv & 1) * 16 + row) << 5;    // lane's B col base in lB

    for (int k0 = 0; k0 < K; k0 += 32) {
        const bool fullK = (k0 + 32 <= K);
        // ---- stage A 32x32; index split keeps global loads coalesced ----
        if (fullM & fullK) {
            #pragma unroll
            for (int it = 0; it < 8; ++it) {
                int idx = tid + it * 128;
                int mm = TA ? (idx & 31) : (idx >> 5);
                int kk = TA ? (idx >> 5) : (idx & 31);
                int gm = bm0 + mm, gk = k0 + kk;
                float v = TA ? A[(long)gk * lda + gm] : A[(long)gm * lda + gk];
                if (SC) v *= scaleK[gk];
                lA[(mm << 5) + kk] = (_Float16)v;
            }
        } else {
            #pragma unroll
            for (int it = 0; it < 8; ++it) {
                int idx = tid + it * 128;
                int mm = TA ? (idx & 31) : (idx >> 5);
                int kk = TA ? (idx >> 5) : (idx & 31);
                int gm = bm0 + mm, gk = k0 + kk;
                bool inb = (gm < M) && (gk < K);
                int cm = inb ? gm : 0, ck = inb ? gk : 0;
                float v = TA ? A[(long)ck * lda + cm] : A[(long)cm * lda + ck];
                if (SC) v *= scaleK[ck];
                lA[(mm << 5) + kk] = (_Float16)(inb ? v : 0.f);
            }
        }
        // ---- stage B 32x32 transposed into lB[n][k] ----
        if (fullN & fullK) {
            #pragma unroll
            for (int it = 0; it < 8; ++it) {
                int idx = tid + it * 128;
                int nn2 = TB ? (idx >> 5) : (idx & 31);
                int kk  = TB ? (idx & 31) : (idx >> 5);
                int gn = bn0 + nn2, gk = k0 + kk;
                float v = TB ? Bm[(long)gn * ldb + gk] : Bm[(long)gk * ldb + gn];
                lB[(nn2 << 5) + kk] = (_Float16)v;
            }
        } else {
            #pragma unroll
            for (int it = 0; it < 8; ++it) {
                int idx = tid + it * 128;
                int nn2 = TB ? (idx >> 5) : (idx & 31);
                int kk  = TB ? (idx & 31) : (idx >> 5);
                int gn = bn0 + nn2, gk = k0 + kk;
                bool inb = (gn < Nn) && (gk < K);
                int cn = inb ? gn : 0, ck = inb ? gk : 0;
                float v = TB ? Bm[(long)cn * ldb + ck] : Bm[(long)ck * ldb + cn];
                lB[(nn2 << 5) + kk] = (_Float16)(inb ? v : 0.f);
            }
        }
        __syncthreads();
        // gather: A lane(row,h): e -> K=16h+e
        //         B lane(col,h): e -> K=16h+e
        v16h av, bv;
        #pragma unroll
        for (int e = 0; e < 16; ++e) av[e]     = lA[aRow + (hh << 4) + e];
        #pragma unroll
        for (int e = 0; e < 16; ++e) bv[e]     = lB[bRow + (hh << 4) + e];

        acc = __builtin_amdgcn_wmma_f32_16x16x32_f16(false, av, false, bv,
                                                     (short)0, acc, false, false);
        __syncthreads();
    }

    // C layout: lane col n = l%16; VGPR r -> m = r + 8*(l/16)
    const int n = n0 + row;
    #pragma unroll
    for (int r = 0; r < 8; ++r) {
        int m = m0 + r + (hh << 3);
        if (m < M && n < Nn) {
            float v = alpha * acc[r];
            if (Cadd) v += addScale * Cadd[(long)m * ldcadd + n];
            if (bias) v += bias[n];
            long o = (long)m * ldc + n;
            if (accumulate) v += C[o];
            if (act == 1)      v = 1.f / (1.f + expf(-v));
            else if (act == 2) v = tanhf(v);
            C[o] = v;
        }
    }
}

// ============================ elementwise ============================
__global__ void k_zero(float* __restrict__ p, int n) {
    int i = blockIdx.x * blockDim.x + threadIdx.x;
    if (i < n) p[i] = 0.f;
}

__global__ void k_build_hi_enc(const float* __restrict__ hist,
                               const float* __restrict__ H,
                               float* __restrict__ hi, int t) {
    int i = blockIdx.x * blockDim.x + threadIdx.x;
    if (i >= BB * NN * XDIM) return;
    int c = i % XDIM; int bn = i / XDIM; int n = bn % NN; int b = bn / NN;
    float v;
    if (c < CIN) v = hist[(((long)b * LL + t) * NN + n) * CIN + c];
    else         v = H[(long)bn * HID + (c - CIN)];
    hi[i] = v;
}

__global__ void k_build_hi_dec(const float* __restrict__ fut,
                               const float* __restrict__ prev,
                               const float* __restrict__ H,
                               float* __restrict__ hi, int t) {
    int i = blockIdx.x * blockDim.x + threadIdx.x;
    if (i >= BB * NN * XDIM) return;
    int c = i % XDIM; int bn = i / XDIM; int n = bn % NN; int b = bn / NN;
    float v;
    if (c == 0)      v = prev[bn];
    else if (c == 1) v = fut[(((long)b * LL + t) * NN + n) * CIN + 1];
    else             v = H[(long)bn * HID + (c - CIN)];
    hi[i] = v;
}

__global__ void k_copy66(const float* __restrict__ src, float* __restrict__ ho) {
    int i = blockIdx.x * blockDim.x + threadIdx.x;
    if (i >= BB * NN * XDIM) return;
    int c = i % XDIM; int bn = i / XDIM;
    ho[(long)bn * DIN + c] = src[i];
}

__global__ void k_nv(const float* __restrict__ f, const float* __restrict__ emb,
                     float* __restrict__ nv) {
    int i = blockIdx.x * blockDim.x + threadIdx.x;
    if (i >= BB * NN * NODE_DIM) return;
    int d = i % NODE_DIM; int n = (i / NODE_DIM) % NN;
    nv[i] = tanhf(TAc * emb[n * NODE_DIM + d] * f[i]);
}

// S = relu(tanh(TA*(a - a^T))) + I
__global__ void k_S(const float* __restrict__ a, float* __restrict__ S) {
    long i = (long)blockIdx.x * blockDim.x + threadIdx.x;
    long tot = (long)BB * NN * NN;
    if (i >= tot) return;
    int j = (int)(i % NN); long bi = i / NN; int ii = (int)(bi % NN); long b = bi / NN;
    long base = b * (long)NN * NN;
    float av = a[base + (long)ii * NN + j] - a[base + (long)j * NN + ii];
    float s = tanhf(TAc * av);
    s = s > 0.f ? s : 0.f;
    if (ii == j) s += 1.f;
    S[i] = s;
}

// invrs[b,i] = 1/row-sum(S), invcs[b,j] = 1/col-sum(S)
__global__ void k_sums(const float* __restrict__ S, float* __restrict__ invrs,
                       float* __restrict__ invcs) {
    int i = blockIdx.x * blockDim.x + threadIdx.x;
    if (i >= BB * NN) return;
    int n = i % NN; int b = i / NN;
    const float* Sb = S + (long)b * NN * NN;
    float rs = 0.f, cs = 0.f;
    for (int j = 0; j < NN; ++j) {
        rs += Sb[(long)n * NN + j];
        cs += Sb[(long)j * NN + n];
    }
    invrs[i] = 1.f / rs;
    invcs[i] = 1.f / cs;
}

__global__ void k_temp(const float* __restrict__ hi, const float* __restrict__ r,
                       const float* __restrict__ H, float* __restrict__ temp) {
    int i = blockIdx.x * blockDim.x + threadIdx.x;
    if (i >= BB * NN * XDIM) return;
    int c = i % XDIM; int bn = i / XDIM;
    temp[i] = (c < CIN) ? hi[i]
                        : r[(long)bn * HID + c - CIN] * H[(long)bn * HID + c - CIN];
}

__global__ void k_updH(const float* __restrict__ z, const float* __restrict__ cn,
                       float* __restrict__ H) {
    int i = blockIdx.x * blockDim.x + threadIdx.x;
    if (i >= BB * NN * HID) return;
    H[i] = z[i] * H[i] + (1.f - z[i]) * cn[i];
}

__global__ void k_final(const float* __restrict__ H, const float* __restrict__ fcw,
                        const float* __restrict__ fcb, float* __restrict__ out,
                        float* __restrict__ prev, int t) {
    int i = blockIdx.x * blockDim.x + threadIdx.x;
    if (i >= BB * NN) return;
    int n = i % NN; int b = i / NN;
    float y = fcb[0];
    for (int k = 0; k < HID; ++k) y += H[(long)i * HID + k] * fcw[k];
    out[((long)b * LL + t) * NN + n] = y;
    prev[i] = y;
}

// ============================ host side ============================
namespace {

struct P {
    const float *hist, *fut, *adj0, *adj1, *emb1, *emb2;
    const float *hw1, *hb1, *hw2, *hb2, *hw3, *hb3, *rw, *rbias, *fcw, *fcb;
    float *out;
    float *hi, *ho, *h1, *h2, *f1, *f2, *nv1, *nv2, *amat, *S;
    float *invrs, *invcs, *z, *r, *cn, *temp, *H, *prev;
};

inline void gemm(hipStream_t st,
                 const float* A, long sA, int lda, int tA,
                 const float* scl, long sScl,
                 const float* Bm, long sB, int ldb, int tB,
                 float* C, long sC, int ldc,
                 const float* Cadd, long sCadd, int ldca, float addS,
                 const float* bias,
                 int M, int Nn, int K, float alpha, int act, int accum, int batch) {
    dim3 g((M + 31) / 32, (Nn + 31) / 32, batch);
    #define LAUNCH(ta, tb, sc) \
        k_gemm<ta, tb, sc><<<g, 128, 0, st>>>(A, sA, lda, scl, sScl, Bm, sB, ldb, \
            C, sC, ldc, Cadd, sCadd, ldca, addS, bias, M, Nn, K, alpha, act, accum)
    int sc = (scl != nullptr);
    if (tA) {
        if (sc) LAUNCH(1, 0, 1);          // adp-folded dynamic adjacency
        else    LAUNCH(1, 0, 0);          // static predefined adjacency
    } else if (tB) {
        LAUNCH(0, 1, 0);                  // nv1 @ nv2^T
    } else {
        if (sc) LAUNCH(0, 0, 1);          // adpT-folded dynamic adjacency
        else    LAUNCH(0, 0, 0);          // MLPs / projections
    }
    #undef LAUNCH
}

inline int nb(long tot) { return (int)((tot + 255) / 256); }

// one rnn-gcn pair: g = gcn(inp, adp, adj0, w[j]) + gcn(inp, adpT, adj1, w[j+1])
// (adp / adpT folded into GEMM via S + invrs/invcs scaling)
void rnn_pair(hipStream_t st, P& p, const float* inp, int j, int act, float* outb) {
    const long sIn = (long)NN * XDIM, sHo = (long)NN * DIN, sAd = (long)NN * NN;
    for (int s = 0; s < 2; ++s) {
        const float* scl = s ? p.invcs : p.invrs;
        int dynTrans = s ? 0 : 1;                 // adp: S^T access; adpT: S access
        const float* pre = s ? p.adj1 : p.adj0;
        // hop1: ho[66:132] = BW*dyn@inp + AW*inp ; then += GW*pre^T@inp
        gemm(st, p.S, sAd, NN, dynTrans, scl, NN, inp, sIn, XDIM, 0,
             p.ho + XDIM, sHo, DIN, inp, sIn, XDIM, AWc, nullptr,
             NN, XDIM, NN, BWc, 0, 0, BB);
        gemm(st, pre, 0, NN, 1, nullptr, 0, inp, sIn, XDIM, 0,
             p.ho + XDIM, sHo, DIN, nullptr, 0, 0, 0.f, nullptr,
             NN, XDIM, NN, GWc, 0, 1, BB);
        // hop2 on h1 = ho[66:132]
        gemm(st, p.S, sAd, NN, dynTrans, scl, NN, p.ho + XDIM, sHo, DIN, 0,
             p.ho + 2 * XDIM, sHo, DIN, inp, sIn, XDIM, AWc, nullptr,
             NN, XDIM, NN, BWc, 0, 0, BB);
        gemm(st, pre, 0, NN, 1, nullptr, 0, p.ho + XDIM, sHo, DIN, 0,
             p.ho + 2 * XDIM, sHo, DIN, nullptr, 0, 0, 0.f, nullptr,
             NN, XDIM, NN, GWc, 0, 1, BB);
        // projection [B*N,198]@[198,64] + bias; second one accumulates + activation
        int wi = j + s;
        gemm(st, p.ho, 0, DIN, 0, nullptr, 0, p.rw + (long)wi * DIN * HID, 0, HID, 0,
             outb, 0, HID, nullptr, 0, 0, 0.f, p.rbias + wi * HID,
             BB * NN, HID, DIN, 1.f, s ? act : 0, s, 1);
    }
}

void run_step(hipStream_t st, P& p, int hy0, int rb, int dec, int t) {
    const long sIn = (long)NN * XDIM, sHo = (long)NN * DIN;
    // hi = concat(x, H)
    if (!dec) k_build_hi_enc<<<nb(BB*NN*XDIM), 256, 0, st>>>(p.hist, p.H, p.hi, t);
    else      k_build_hi_dec<<<nb(BB*NN*XDIM), 256, 0, st>>>(p.fut, p.prev, p.H, p.hi, t);
    k_copy66<<<nb(BB*NN*XDIM), 256, 0, st>>>(p.hi, p.ho);

    // ---- hyper GCNs: shared diffusion per adjacency, 2 MLP chains each ----
    for (int a = 0; a < 2; ++a) {
        const float* adj = a ? p.adj1 : p.adj0;
        gemm(st, adj, 0, NN, 1, nullptr, 0, p.hi, sIn, XDIM, 0,
             p.ho + XDIM, sHo, DIN, p.hi, sIn, XDIM, AWc, nullptr,
             NN, XDIM, NN, GWc, 0, 0, BB);
        gemm(st, adj, 0, NN, 1, nullptr, 0, p.ho + XDIM, sHo, DIN, 0,
             p.ho + 2 * XDIM, sHo, DIN, p.hi, sIn, XDIM, AWc, nullptr,
             NN, XDIM, NN, GWc, 0, 0, BB);
        for (int c2 = 0; c2 < 2; ++c2) {
            int wi = hy0 + 2 * c2 + a;
            float* fout = c2 ? p.f2 : p.f1;
            gemm(st, p.ho, 0, DIN, 0, nullptr, 0,
                 p.hw1 + (long)wi * DIN * HYP_DIM, 0, HYP_DIM, 0,
                 p.h1, 0, HYP_DIM, nullptr, 0, 0, 0.f, p.hb1 + wi * HYP_DIM,
                 BB * NN, HYP_DIM, DIN, 1.f, 1, 0, 1);
            gemm(st, p.h1, 0, HYP_DIM, 0, nullptr, 0,
                 p.hw2 + (long)wi * HYP_DIM * MID_DIM, 0, MID_DIM, 0,
                 p.h2, 0, MID_DIM, nullptr, 0, 0, 0.f, p.hb2 + wi * MID_DIM,
                 BB * NN, MID_DIM, HYP_DIM, 1.f, 1, 0, 1);
            gemm(st, p.h2, 0, MID_DIM, 0, nullptr, 0,
                 p.hw3 + (long)wi * MID_DIM * NODE_DIM, 0, NODE_DIM, 0,
                 fout, 0, NODE_DIM, nullptr, 0, 0, 0.f, p.hb3 + wi * NODE_DIM,
                 BB * NN, NODE_DIM, MID_DIM, 1.f, 0, a, 1);
        }
    }
    // node vectors + dynamic adjacency
    k_nv<<<nb(BB*NN*NODE_DIM), 256, 0, st>>>(p.f1, p.emb1, p.nv1);
    k_nv<<<nb(BB*NN*NODE_DIM), 256, 0, st>>>(p.f2, p.emb2, p.nv2);
    gemm(st, p.nv1, (long)NN * NODE_DIM, NODE_DIM, 0, nullptr, 0,
         p.nv2, (long)NN * NODE_DIM, NODE_DIM, 1,
         p.amat, (long)NN * NN, NN, nullptr, 0, 0, 0.f, nullptr,
         NN, NN, NODE_DIM, 1.f, 0, 0, BB);
    k_S<<<nb((long)BB*NN*NN), 256, 0, st>>>(p.amat, p.S);
    k_sums<<<nb(BB*NN), 256, 0, st>>>(p.S, p.invrs, p.invcs);

    // ---- GRU gates ----
    rnn_pair(st, p, p.hi, rb + 0, 1, p.z);      // z = sigmoid(g(hi,0))
    rnn_pair(st, p, p.hi, rb + 2, 1, p.r);      // r = sigmoid(g(hi,2))
    k_temp<<<nb(BB*NN*XDIM), 256, 0, st>>>(p.hi, p.r, p.H, p.temp);
    k_copy66<<<nb(BB*NN*XDIM), 256, 0, st>>>(p.temp, p.ho);
    rnn_pair(st, p, p.temp, rb + 4, 2, p.cn);   // Cn = tanh(g(temp,4))
    k_updH<<<nb(BB*NN*HID), 256, 0, st>>>(p.z, p.cn, p.H);

    if (dec)
        k_final<<<nb(BB*NN), 256, 0, st>>>(p.H, p.fcw, p.fcb, p.out, p.prev, t);
}

} // namespace

extern "C" void kernel_launch(void* const* d_in, const int* in_sizes, int n_in,
                              void* d_out, int out_size, void* d_ws, size_t ws_size,
                              hipStream_t stream) {
    P p;
    p.hist = (const float*)d_in[0];
    p.fut  = (const float*)d_in[1];
    p.adj0 = (const float*)d_in[2];
    p.adj1 = (const float*)d_in[3];
    p.emb1 = (const float*)d_in[4];
    p.emb2 = (const float*)d_in[5];
    p.hw1  = (const float*)d_in[6];  p.hb1 = (const float*)d_in[7];
    p.hw2  = (const float*)d_in[8];  p.hb2 = (const float*)d_in[9];
    p.hw3  = (const float*)d_in[10]; p.hb3 = (const float*)d_in[11];
    p.rw   = (const float*)d_in[12]; p.rbias = (const float*)d_in[13];
    p.fcw  = (const float*)d_in[14]; p.fcb = (const float*)d_in[15];
    p.out  = (float*)d_out;

    // workspace carve-up (~59 MB of floats)
    float* w = (float*)d_ws;
    p.hi   = w; w += (long)BB * NN * XDIM;
    p.ho   = w; w += (long)BB * NN * DIN;
    p.h1   = w; w += (long)BB * NN * HYP_DIM;
    p.h2   = w; w += (long)BB * NN * MID_DIM;
    p.f1   = w; w += (long)BB * NN * NODE_DIM;
    p.f2   = w; w += (long)BB * NN * NODE_DIM;
    p.nv1  = w; w += (long)BB * NN * NODE_DIM;
    p.nv2  = w; w += (long)BB * NN * NODE_DIM;
    p.amat = w; w += (long)BB * NN * NN;
    p.S    = w; w += (long)BB * NN * NN;
    p.invrs = w; w += (long)BB * NN;
    p.invcs = w; w += (long)BB * NN;
    p.z    = w; w += (long)BB * NN * HID;
    p.r    = w; w += (long)BB * NN * HID;
    p.cn   = w; w += (long)BB * NN * HID;
    p.temp = w; w += (long)BB * NN * XDIM;
    p.H    = w; w += (long)BB * NN * HID;
    p.prev = w; w += (long)BB * NN;

    // deterministic init each call
    k_zero<<<nb(BB*NN*HID), 256, 0, stream>>>(p.H, BB * NN * HID);
    k_zero<<<nb(BB*NN), 256, 0, stream>>>(p.prev, BB * NN);

    // encoder: hyper weights 0..3, rnn weights 0..5
    for (int t = 0; t < LL; ++t) run_step(stream, p, 0, 0, 0, t);
    // decoder: hyper weights 4..7, rnn weights 6..11, task_level == 12
    for (int t = 0; t < LL; ++t) run_step(stream, p, 4, 6, 1, t);
}
